// LinearSpatCosSimTempConv_41686952575616
// MI455X (gfx1250) — compile-verified, matrix operand order checked by reading, not run on hardware
//
#include <hip/hip_runtime.h>
#include <hip/hip_bf16.h>

// ---------------------------------------------------------------------------
// Fused LinearSpat + cosine-sim temporal conv for MI455X (gfx1250, wave32).
//
//  Stage 1: spat[o,t] = spat_w[o,:] . x[b,:,t]        -> WMMA bf16 GEMM
//  Stage 2: conv[o,t] = sum_k w[o,k]*spat[o,t+k]      -> banded-Toeplitz WMMA
//           sq[o,t]   = sum_k spat[o,t+k]^2           -> same band, ones B, A^2
//  out = conv * sqrt(K) / (||w_o|| * sqrt(sq))
//
// All fragment halves are K-contiguous octets per the ISA 16-bit layout, so
// they load as single aligned ds_load_b128 / global_load_b128.
// ---------------------------------------------------------------------------

typedef __attribute__((ext_vector_type(16))) __bf16 v16bf;
typedef __attribute__((ext_vector_type(8)))  __bf16 bf16x8;
typedef __attribute__((ext_vector_type(8)))  float  v8f;

#define O_CH   128
#define CIN    64
#define T_IN   4096
#define KW     64
#define T_OUT  (T_IN - KW + 1)      // 4033
#define TILE_T 256                  // outputs per workgroup tile
#define XS_W   320                  // staged input window (TILE_T + KW - 1 -> 320)
#define XT_S   72                   // xT row stride (16B aligned, bank-spread)
#define SP_W   336                  // spat row stride: covers j index up to 335
#define WP_W   112                  // padded filter row: 16 zeros | 64 w | 32 zeros
#define WP_PAD 16
#define WGSIZE 256                  // 8 wave32

union VB16 { v16bf v; bf16x8 h[2]; __bf16 e[16]; };

// K-octet base for fragment element pair v (ISA 16-bit A/B layout):
// VGPR 0..3 hold K = 8*half + {0..7}; VGPR 4..7 hold K = 16 + 8*half + {0..7}
__device__ __forceinline__ int kbase(int v, int hh) {
    return (v < 4 ? 2 * v : 16 + 2 * (v - 4)) + 8 * hh;
}

// 8 consecutive f32 from global -> bf16 octet (2x global_load_b128 + pk cvt)
__device__ __forceinline__ bf16x8 cvt8(const float* p) {
    float4 lo = *(const float4*)p;
    float4 hi = *(const float4*)(p + 4);
    bf16x8 r;
    r[0] = (__bf16)lo.x; r[1] = (__bf16)lo.y; r[2] = (__bf16)lo.z; r[3] = (__bf16)lo.w;
    r[4] = (__bf16)hi.x; r[5] = (__bf16)hi.y; r[6] = (__bf16)hi.z; r[7] = (__bf16)hi.w;
    return r;
}

__global__ __launch_bounds__(WGSIZE, 2)
void lin_spat_cossim_kernel(const float* __restrict__ x,
                            const float* __restrict__ conv_w,
                            const float* __restrict__ spat_w,
                            float* __restrict__ out) {
    // xT lives only in phase 1, wconvP only in phase 2 -> overlay them.
    __shared__ union {
        __align__(16) __bf16 xT[XS_W][XT_S];       // 46080 B  x tile, transposed
        __align__(16) __bf16 wconvP[O_CH][WP_W];   // 28672 B  zero-padded filters
    } smem;
    __shared__ float s_rnorm[O_CH];                             // sqrt(K)/||w_o||
    __shared__ __align__(16) __bf16 s_sp[O_CH][SP_W];           // 86016 B spat tile

    const int tid  = threadIdx.x;
    const int wave = tid >> 5;
    const int lane = tid & 31;
    const int hh   = lane >> 4;        // lane half (0: lanes 0-15, 1: 16-31)
    const int ln   = lane & 15;
    const int t0   = blockIdx.x * TILE_T;
    const int b    = blockIdx.y;

    // ---------------- Phase 0: stage x tile (transposed, bf16) -------------
    const float* xb = x + (size_t)b * CIN * T_IN;
    if (t0 + XS_W <= T_IN) {           // fast path: full window in range
        for (int i = tid; i < CIN * (XS_W / 4); i += WGSIZE) {
            int r = i / (XS_W / 4);
            int c = (i - r * (XS_W / 4)) * 4;
            const float4 vx = *(const float4*)(xb + (size_t)r * T_IN + t0 + c);
            smem.xT[c + 0][r] = (__bf16)vx.x;
            smem.xT[c + 1][r] = (__bf16)vx.y;
            smem.xT[c + 2][r] = (__bf16)vx.z;
            smem.xT[c + 3][r] = (__bf16)vx.w;
        }
    } else {                           // last t tile: guarded scalar path
        for (int i = tid; i < CIN * XS_W; i += WGSIZE) {
            int r = i / XS_W, c = i - r * XS_W;
            int t = t0 + c;
            smem.xT[c][r] = (t < T_IN) ? (__bf16)xb[(size_t)r * T_IN + t]
                                       : (__bf16)0.0f;
        }
    }
    // zero spat padding columns [320,336) so padded-K WMMA terms are 0*0
    for (int i = tid; i < O_CH * (SP_W - XS_W); i += WGSIZE) {
        int r = i / (SP_W - XS_W), c = XS_W + (i - r * (SP_W - XS_W));
        s_sp[r][c] = (__bf16)0.0f;
    }
    if (tid < O_CH) {
        float s = 0.0f;
        #pragma unroll 8
        for (int k = 0; k < KW; ++k) { float w = conv_w[tid * KW + k]; s += w * w; }
        s_rnorm[tid] = 8.0f * rsqrtf(s);    // sqrt(K) = 8
    }
    __syncthreads();

    // ---------------- Phase 1: spat = spat_w @ x (WMMA bf16 GEMM) ----------
    {
        const int m0 = wave * 16;                       // this wave's O block
        const float* wsrow = spat_w + (size_t)(m0 + ln) * CIN;
        VB16 a0, a1;                   // A frags (K-steps over Cin), from global
        a0.h[0] = cvt8(wsrow +      8 * hh);
        a0.h[1] = cvt8(wsrow + 16 + 8 * hh);
        a1.h[0] = cvt8(wsrow + 32 + 8 * hh);
        a1.h[1] = cvt8(wsrow + 48 + 8 * hh);

        for (int nt = 0; nt < XS_W / 16; ++nt) {        // 20 N-tiles of t
            const int tb = nt * 16;
            const __bf16* xrow = &smem.xT[tb + ln][0];  // column t = tb+ln
            VB16 b0, b1;                                // 4x ds_load_b128
            b0.h[0] = *(const bf16x8*)(xrow +      8 * hh);
            b0.h[1] = *(const bf16x8*)(xrow + 16 + 8 * hh);
            b1.h[0] = *(const bf16x8*)(xrow + 32 + 8 * hh);
            b1.h[1] = *(const bf16x8*)(xrow + 48 + 8 * hh);
            v8f d = {};
            d = __builtin_amdgcn_wmma_f32_16x16x32_bf16(false, a0.v, false, b0.v,
                                                        (short)0, d, false, false);
            d = __builtin_amdgcn_wmma_f32_16x16x32_bf16(false, a1.v, false, b1.v,
                                                        (short)0, d, false, false);
            #pragma unroll
            for (int g = 0; g < 8; ++g)                 // D: m = g + 8*hh, n = ln
                s_sp[m0 + g + 8 * hh][tb + ln] = (__bf16)d[g];
        }
    }
    __syncthreads();

    // ---------------- Phase 1b: overlay zero-padded filters over xT --------
    for (int i = tid; i < O_CH * WP_W; i += WGSIZE) {
        int o = i / WP_W, c = i - o * WP_W;
        int k = c - WP_PAD;
        ((__bf16*)smem.wconvP)[i] = (k >= 0 && k < KW) ? (__bf16)conv_w[o * KW + k]
                                                       : (__bf16)0.0f;
    }
    __syncthreads();

    // ---------------- Phase 2: banded-Toeplitz WMMA conv + window norm -----
    // out[16m+n] = sum_j spat[16m+j] * band[j,n], band[j,n] = w[j-n], j in [0,96)
    VB16 ones[3];                       // all-ones band for sum-of-squares
    #pragma unroll
    for (int c = 0; c < 3; ++c)
        #pragma unroll
        for (int v = 0; v < 8; ++v) {
            int kb = kbase(v, hh);
            #pragma unroll
            for (int jj = 0; jj < 2; ++jj) {
                int dlt = 32 * c + kb + jj - ln;
                ones[c].e[2*v+jj] = (dlt >= 0 && dlt < KW) ? (__bf16)1.0f
                                                           : (__bf16)0.0f;
            }
        }

    float* outb = out + (size_t)b * O_CH * T_OUT;
    #pragma unroll 2
    for (int oi = 0; oi < O_CH / 8; ++oi) {
        const int o = wave * 16 + oi;          // each wave covers 16 channels

        // banded filter B frags: single per-lane base, immediate offsets only
        const __bf16* wp = &smem.wconvP[o][WP_PAD - ln];
        VB16 bw[3];
        #pragma unroll
        for (int c = 0; c < 3; ++c)
            #pragma unroll
            for (int v = 0; v < 8; ++v) {
                int kb = kbase(v, hh);
                bw[c].e[2*v]   = wp[32 * c + kb];
                bw[c].e[2*v+1] = wp[32 * c + kb + 1];
            }

        const __bf16* sprow = &s_sp[o][16 * ln];
        v8f dc = {}, ds2 = {};
        #pragma unroll
        for (int c = 0; c < 3; ++c) {
            VB16 a, a2;                         // A rows = 16 t-blocks of chan o
            a.h[0] = *(const bf16x8*)(sprow + 32 * c +      8 * hh);
            a.h[1] = *(const bf16x8*)(sprow + 32 * c + 16 + 8 * hh);
            a2.v = a.v * a.v;                   // packed bf16 squares (v_pk_mul_bf16)
            dc  = __builtin_amdgcn_wmma_f32_16x16x32_bf16(false, a.v,  false, bw[c].v,
                                                          (short)0, dc,  false, false);
            ds2 = __builtin_amdgcn_wmma_f32_16x16x32_bf16(false, a2.v, false, ones[c].v,
                                                          (short)0, ds2, false, false);
        }

        const float rn = s_rnorm[o];
        #pragma unroll
        for (int g = 0; g < 8; ++g) {
            int m = g + 8 * hh;
            int t = t0 + 16 * m + ln;
            if (t < T_OUT)
                outb[(size_t)o * T_OUT + t] = dc[g] * rn * rsqrtf(ds2[g]);
        }
    }
}

extern "C" void kernel_launch(void* const* d_in, const int* in_sizes, int n_in,
                              void* d_out, int out_size, void* d_ws, size_t ws_size,
                              hipStream_t stream) {
    const float* x      = (const float*)d_in[0];    // [64,64,4096]
    const float* conv_w = (const float*)d_in[1];    // [128,64]
    const float* spat_w = (const float*)d_in[2];    // [128,64]
    float*       out    = (float*)d_out;            // [64,128,4033]

    dim3 grid((T_OUT + TILE_T - 1) / TILE_T, 64);   // 16 t-tiles x 64 batches
    lin_spat_cossim_kernel<<<grid, WGSIZE, 0, stream>>>(x, conv_w, spat_w, out);
}
